// _ParoQuantOptimLinear_13005160972529
// MI455X (gfx1250) — matrix operand order repaired and checked by reading, not compiled, not run on hardware
//
#include <hip/hip_runtime.h>
#include <hip/hip_bf16.h>
#include <cmath>

// ---------------------------------------------------------------------------
// Problem constants (match the reference)
// ---------------------------------------------------------------------------
#define IN_F     4096
#define OUT_F    4096
#define N_TOK    8192
#define GROUPSZ  128
#define N_STAGES 4
#define QMAXF    15.0f
#define PAIRS_PER_STAGE (IN_F / 2)          // 2048
#define GROUPS_PER_ROW  (IN_F / GROUPSZ)    // 32

// GEMM tiling: block = 256 threads = 8 waves (4x2 wave grid),
// block tile 256x128, wave tile 64x64 (4x4 WMMA tiles), K step 32.
#define BM 256
#define BN 128
#define BK 32
#define NKSTEPS (IN_F / BK)                 // 128
#define A_TILE_HALVES (BM * BK)             // 8192 (16 KB)
#define B_TILE_HALVES (BN * BK)             // 4096 (8 KB)
#define BUF_HALVES (A_TILE_HALVES + B_TILE_HALVES)

typedef __attribute__((ext_vector_type(16))) _Float16 v16h;
typedef __attribute__((ext_vector_type(8)))  _Float16 v8h;
typedef __attribute__((ext_vector_type(8)))  float    v8f;

typedef __attribute__((address_space(3))) char* lchar3_t;

// ---------------------------------------------------------------------------
// CDNA5 async global->LDS copy (ASYNCcnt-tracked), via inline asm.
//   lds : 32-bit LDS byte address (per lane)
//   g   : 64-bit global address (per lane)
// ---------------------------------------------------------------------------
__device__ __forceinline__ void async_copy_b128(uint32_t lds, const void* g) {
    asm volatile("global_load_async_to_lds_b128 %0, %1, off"
                 :: "v"(lds), "v"((unsigned long long)(uintptr_t)g)
                 : "memory");
}
__device__ __forceinline__ void wait_async0() {
    asm volatile("s_wait_asynccnt 0x0" ::: "memory");
}

// ---------------------------------------------------------------------------
// Kernel 0: cos/sin tables for the rotation angles (shared across all rows)
// ---------------------------------------------------------------------------
__global__ void trig_tables_kernel(const float* __restrict__ theta,
                                   float* __restrict__ ct,
                                   float* __restrict__ st) {
    int i = blockIdx.x * blockDim.x + threadIdx.x;   // 0 .. 8191
    if (i < N_STAGES * PAIRS_PER_STAGE) {
        float t = theta[i];
        ct[i] = cosf(t);
        st[i] = sinf(t);
    }
}

// ---------------------------------------------------------------------------
// Kernel 1: convert x (fp32) -> f16 workspace copy (row-major, K contiguous)
// ---------------------------------------------------------------------------
__global__ void cvt_x_kernel(const float* __restrict__ X,
                             _Float16* __restrict__ Xh) {
    size_t base = ((size_t)blockIdx.x * blockDim.x + threadIdx.x) * 8;
    float4 a = *(const float4*)(X + base);
    float4 b = *(const float4*)(X + base + 4);
    v8h h;
    h[0] = (_Float16)a.x; h[1] = (_Float16)a.y;
    h[2] = (_Float16)a.z; h[3] = (_Float16)a.w;
    h[4] = (_Float16)b.x; h[5] = (_Float16)b.y;
    h[6] = (_Float16)b.z; h[7] = (_Float16)b.w;
    *(v8h*)(Xh + base) = h;
}

// ---------------------------------------------------------------------------
// Kernel 2: per-output-row weight pipeline in LDS (one block per row).
// ---------------------------------------------------------------------------
__global__ __launch_bounds__(256) void prep_w_kernel(
    const float* __restrict__ W,
    const float* __restrict__ cs,
    const float* __restrict__ qs,
    const float* __restrict__ qzp,
    const int*   __restrict__ pairs,
    const float* __restrict__ ct,
    const float* __restrict__ st,
    _Float16*    __restrict__ Wh) {

    __shared__ float row[IN_F];
    const int o   = blockIdx.x;
    const int tid = threadIdx.x;
    const float* wr = W + (size_t)o * IN_F;

    for (int k = tid; k < IN_F; k += 256) row[k] = wr[k] * cs[k];
    __syncthreads();

    // forward rotations: xi' = xi*c + xj*s ; xj' = -xi*s + xj*c
    for (int r = 0; r < N_STAGES; ++r) {
        const int* pr = pairs + r * IN_F;
        const float* cr = ct + r * PAIRS_PER_STAGE;
        const float* sr = st + r * PAIRS_PER_STAGE;
        for (int p = tid; p < PAIRS_PER_STAGE; p += 256) {
            int g = p >> 6, ph = p & 63, base = g << 7;
            int i = pr[base + 2 * ph]     + base;
            int j = pr[base + 2 * ph + 1] + base;
            float c = cr[p], s = sr[p];
            float xi = row[i], xj = row[j];
            row[i] =  xi * c + xj * s;
            row[j] = -xi * s + xj * c;
        }
        __syncthreads();
    }

    // 4-bit affine quant-dequant per 128-group (RNE rounding == jnp.round)
    for (int k = tid; k < IN_F; k += 256) {
        int g = k >> 7;
        float s = qs[(size_t)o * GROUPS_PER_ROW + g];
        s = fminf(fmaxf(s, 1e-5f), 1e5f);
        float rzp = fminf(fmaxf(-rintf(qzp[(size_t)o * GROUPS_PER_ROW + g]), 0.0f), QMAXF);
        float q = rintf(row[k] / s) + rzp;
        q = fminf(fmaxf(q, 0.0f), QMAXF);
        row[k] = (q - rzp) * s;
    }
    __syncthreads();

    // inverse rotations (reversed stage order, -theta)
    for (int r = N_STAGES - 1; r >= 0; --r) {
        const int* pr = pairs + r * IN_F;
        const float* cr = ct + r * PAIRS_PER_STAGE;
        const float* sr = st + r * PAIRS_PER_STAGE;
        for (int p = tid; p < PAIRS_PER_STAGE; p += 256) {
            int g = p >> 6, ph = p & 63, base = g << 7;
            int i = pr[base + 2 * ph]     + base;
            int j = pr[base + 2 * ph + 1] + base;
            float c = cr[p], s = sr[p];
            float xi = row[i], xj = row[j];
            row[i] = xi * c - xj * s;
            row[j] = xi * s + xj * c;
        }
        __syncthreads();
    }

    for (int k = tid; k < IN_F; k += 256)
        Wh[(size_t)o * IN_F + k] = (_Float16)(row[k] / cs[k]);
}

// ---------------------------------------------------------------------------
// Kernel 3: WMMA GEMM  out = Xh @ Wh^T + bias
//
// LDS tiles are written directly in the ISA fragment layouts (7.12.2):
//   A (16-bit 16x32): lane<16 holds M=lane, halves 0-7 = K0..7, 8-15 = K16..23;
//                     lane>=16 holds M=lane-16, K8..15 / K24..31.
//   B (32x16):        lane<16 holds N=lane, halves 0-7 = K0..7, 8-15 = K16..23;
//                     lane>=16 holds N=lane-16, K8..15 / K24..31.
// so each fragment read is two aligned ds_load_b128, and each staging run is
// one 16-byte global->LDS async copy (ASYNCcnt tracked, no VGPR data).
// ---------------------------------------------------------------------------
__global__ __launch_bounds__(256, 1) void gemm_kernel(
    const _Float16* __restrict__ Xh,
    const _Float16* __restrict__ Wh,
    const float*    __restrict__ bias,
    float*          __restrict__ out) {

    __shared__ __align__(16) _Float16 smem[2 * BUF_HALVES];   // 48 KB

    const int tid  = threadIdx.x;
    const int lane = tid & 31;
    const int wid  = tid >> 5;
    const int wm   = wid >> 1;        // 0..3 : 64-row strip of the 256-row tile
    const int wn   = wid & 1;         // 0..1 : 64-col strip of the 128-col tile
    const int bx   = blockIdx.x;      // N block (0..31)
    const int by   = blockIdx.y;      // M block (0..31)

    // ---- per-thread staging addresses (constant across K steps) ----
    // A: 1024 runs of 16B (row 0..255 x K-octet 0..3), 4 per thread
    // B:  512 runs of 16B (col 0..127 x K-octet 0..3), 2 per thread
    size_t   gA[4], gB[2];            // byte offsets from Xh / Wh
    uint32_t lA[4], lB[2];            // byte offsets within one LDS buffer
    #pragma unroll
    for (int t = 0; t < 4; ++t) {
        int run = tid + t * 256;
        int m = run >> 2, g = run & 3;
        gA[t] = (((size_t)(by * BM + m)) * IN_F + g * 8) * sizeof(_Float16);
        lA[t] = ((((m >> 4) * 32 + (m & 15) + ((g & 1) << 4)) * 16 + (g >> 1) * 8))
                * sizeof(_Float16);
    }
    #pragma unroll
    for (int t = 0; t < 2; ++t) {
        int run = tid + t * 256;
        int n = run >> 2, g = run & 3;
        gB[t] = (((size_t)(bx * BN + n)) * IN_F + g * 8) * sizeof(_Float16);
        lB[t] = (A_TILE_HALVES
                 + (((n >> 4) * 32 + (n & 15) + ((g & 1) << 4)) * 16 + (g >> 1) * 8))
                * sizeof(_Float16);
    }

    // 32-bit LDS byte address of the staging buffers (addrspace(3) offset)
    const uint32_t ldsBase = (uint32_t)(uintptr_t)(lchar3_t)&smem[0];

    auto stage = [&](int kb, int buf) {
        const size_t   kOff   = (size_t)kb * BK * sizeof(_Float16);     // 64 B
        const uint32_t bufOff = ldsBase + (uint32_t)buf * BUF_HALVES * sizeof(_Float16);
        #pragma unroll
        for (int t = 0; t < 4; ++t)
            async_copy_b128(bufOff + lA[t], (const char*)Xh + gA[t] + kOff);
        #pragma unroll
        for (int t = 0; t < 2; ++t)
            async_copy_b128(bufOff + lB[t], (const char*)Wh + gB[t] + kOff);
    };

    v8f acc[4][4] = {};

    stage(0, 0);
    wait_async0();
    __syncthreads();

    for (int kb = 0; kb < NKSTEPS; ++kb) {
        const int cur = kb & 1;
        if (kb + 1 < NKSTEPS) stage(kb + 1, cur ^ 1);

        const _Float16* sA = smem + cur * BUF_HALVES;
        const _Float16* sB = sA + A_TILE_HALVES;

        v16h a[4], b[4];
        #pragma unroll
        for (int i = 0; i < 4; ++i)
            a[i] = *(const v16h*)(sA + (((wm * 4 + i) * 32 + lane) * 16));
        #pragma unroll
        for (int j = 0; j < 4; ++j)
            b[j] = *(const v16h*)(sB + (((wn * 4 + j) * 32 + lane) * 16));

        #pragma unroll
        for (int i = 0; i < 4; ++i)
            #pragma unroll
            for (int j = 0; j < 4; ++j)
                acc[i][j] = __builtin_amdgcn_wmma_f32_16x16x32_f16(
                    false, a[i], false, b[j], (short)0, acc[i][j], false, false);

        wait_async0();          // our async writes to buf^1 have landed
        __syncthreads();        // everyone's writes visible before next reads
    }

    // Epilogue: C/D layout (VGPR r: M = r + 8*(lane>=16), N = lane%16)
    const int colBase = bx * BN + wn * 64 + (lane & 15);
    const int rowBase = by * BM + wm * 64 + ((lane >> 4) << 3);
    #pragma unroll
    for (int j = 0; j < 4; ++j) {
        const int col = colBase + j * 16;
        const float bv = bias[col];
        #pragma unroll
        for (int i = 0; i < 4; ++i) {
            #pragma unroll
            for (int r = 0; r < 8; ++r) {
                int row = rowBase + i * 16 + r;
                out[(size_t)row * OUT_F + col] = acc[i][j][r] + bv;
            }
        }
    }
}

// ---------------------------------------------------------------------------
// Host launcher.
// Workspace: [0,32MB) Wh f16 | [32MB,96MB) Xh f16 | then 64KB trig tables
// ---------------------------------------------------------------------------
extern "C" void kernel_launch(void* const* d_in, const int* in_sizes, int n_in,
                              void* d_out, int out_size, void* d_ws, size_t ws_size,
                              hipStream_t stream) {
    const float* x      = (const float*)d_in[0];
    const float* weight = (const float*)d_in[1];
    const float* bias   = (const float*)d_in[2];
    const float* cs     = (const float*)d_in[3];
    const float* theta  = (const float*)d_in[4];
    const float* qs     = (const float*)d_in[5];
    const float* qzp    = (const float*)d_in[6];
    const int*   pairs  = (const int*)d_in[7];
    // d_in[8] = theta_mask (unused by the reference computation)
    float* out = (float*)d_out;

    char* ws = (char*)d_ws;
    const size_t WH_BYTES = (size_t)OUT_F * IN_F * sizeof(_Float16);   // 32 MiB
    const size_t XH_BYTES = (size_t)N_TOK * IN_F * sizeof(_Float16);   // 64 MiB
    _Float16* Wh = (_Float16*)ws;
    _Float16* Xh = (_Float16*)(ws + WH_BYTES);
    float*    ct = (float*)(ws + WH_BYTES + XH_BYTES);
    float*    st = ct + N_STAGES * PAIRS_PER_STAGE;

    trig_tables_kernel<<<(N_STAGES * PAIRS_PER_STAGE + 255) / 256, 256, 0, stream>>>(
        theta, ct, st);

    {
        size_t n = (size_t)N_TOK * IN_F;
        cvt_x_kernel<<<(unsigned)(n / (256 * 8)), 256, 0, stream>>>(x, Xh);
    }

    prep_w_kernel<<<OUT_F, 256, 0, stream>>>(weight, cs, qs, qzp, pairs, ct, st, Wh);

    dim3 grid(OUT_F / BN, N_TOK / BM);   // (32, 32)
    gemm_kernel<<<grid, 256, 0, stream>>>(Xh, Wh, bias, out);
}
